// SpatioTemporalAttention_55035710931606
// MI455X (gfx1250) — matrix operand order repaired and checked by reading, not compiled, not run on hardware
//
#include <hip/hip_runtime.h>

typedef _Float16 f16;
typedef __attribute__((ext_vector_type(16))) _Float16 v16h;
typedef __attribute__((ext_vector_type(8)))  float    v8f;

#define IN_C      256
#define OUT_C     256
#define NUM_HEADS 8
#define HEAD_C    32
#define NBATCH    16
#define LSEQ      2000
#define QKV_COLS  768

// ---------------------------------------------------------------------------
// Kernel 1: qkv = x @ w_qkv, scatter to head-major f16 Q/K/V ([N,H,L,hd]).
// One wave computes a 16x32 output slab (two 16x16 WMMA tiles sharing the A
// fragment). Q is pre-scaled by 1/sqrt(OUT_C).
// ---------------------------------------------------------------------------
__global__ __launch_bounds__(128)
void qkv_gemm_kernel(const float* __restrict__ x, const float* __restrict__ w_qkv,
                     f16* __restrict__ qws, f16* __restrict__ kws, f16* __restrict__ vws)
{
    const int lane    = threadIdx.x & 31;
    const int wave    = threadIdx.x >> 5;
    const int colBase = blockIdx.x * 32;         // 0,32,..,736
    const int rowTile = blockIdx.y * 4 + wave;   // 0..1999
    const int rowBase = rowTile * 16;
    const int mrow    = lane & 15;
    const int laneHi  = lane >> 4;

    v8f acc0 = {}, acc1 = {};
    #pragma unroll
    for (int k0 = 0; k0 < IN_C; k0 += 32) {
        v16h a, b0, b1;
        const float* arow = x + (size_t)(rowBase + mrow) * IN_C + k0;
        #pragma unroll
        for (int v = 0; v < 8; ++v) {
            // A (16x32, 16-bit): lanes 0-15 hold K=0..15, lanes 16-31 hold K=16..31
            int k = laneHi * 16 + 2 * v;
            a[2 * v]     = (f16)arow[k];
            a[2 * v + 1] = (f16)arow[k + 1];
        }
        #pragma unroll
        for (int v = 0; v < 8; ++v) {
            // B (32x16, 16-bit): lanes 0-15 hold K=0..15, lanes 16-31 hold K=16..31
            int k = laneHi * 16 + 2 * v;
            const float* brow0 = w_qkv + (size_t)(k0 + k)     * QKV_COLS + colBase + mrow;
            const float* brow1 = w_qkv + (size_t)(k0 + k + 1) * QKV_COLS + colBase + mrow;
            b0[2 * v]     = (f16)brow0[0];
            b0[2 * v + 1] = (f16)brow1[0];
            b1[2 * v]     = (f16)brow0[16];
            b1[2 * v + 1] = (f16)brow1[16];
        }
        acc0 = __builtin_amdgcn_wmma_f32_16x16x32_f16(false, a, false, b0,
                                                      (short)0, acc0, false, false);
        acc1 = __builtin_amdgcn_wmma_f32_16x16x32_f16(false, a, false, b1,
                                                      (short)0, acc1, false, false);
    }

    const float qscale = 0.0625f;  // 1/sqrt(OUT_C) folded into Q
    #pragma unroll
    for (int t = 0; t < 2; ++t) {
        v8f acc = t ? acc1 : acc0;
        #pragma unroll
        for (int r = 0; r < 8; ++r) {
            int m   = r + laneHi * 8;            // D layout: M = vgpr + 8*(lane>=16)
            int row = rowBase + m;               // flat n*L + l
            int n   = row / LSEQ, l = row % LSEQ;
            int c   = colBase + t * 16 + mrow;   // 0..767
            int head = c / (3 * HEAD_C), within = c % (3 * HEAD_C);
            int which = within / HEAD_C, d = within % HEAD_C;
            size_t off = ((size_t)(n * NUM_HEADS + head) * LSEQ + l) * HEAD_C + d;
            float val = acc[r];
            if (which == 0)      qws[off] = (f16)(val * qscale);
            else if (which == 1) kws[off] = (f16)val;
            else                 vws[off] = (f16)val;
        }
    }
}

// ---------------------------------------------------------------------------
// Kernel 2: streaming (flash) attention. Block = 4 waves, each wave owns a
// 16-query tile of the same (n,h); K/V 32-key tiles staged in LDS via
// GLOBAL_LOAD_ASYNC_TO_LDS_B128 (ASYNCcnt) and shared across waves.
// Per key iteration: 2 WMMAs for S, online softmax, 2 WMMAs for O += P*V.
// ---------------------------------------------------------------------------
__global__ __launch_bounds__(128)
void attn_kernel(const f16* __restrict__ qws, const f16* __restrict__ kws,
                 const f16* __restrict__ vws, f16* __restrict__ aws)
{
    __shared__ f16 sK[32][HEAD_C];     // [key-in-tile][d]
    __shared__ f16 sV[32][HEAD_C];
    __shared__ f16 sP[4][16][32];      // per-wave P tile: [m][key-in-tile]

    const int lane   = threadIdx.x & 31;
    const int wave   = threadIdx.x >> 5;
    const int h      = blockIdx.y;
    const int n      = blockIdx.z;
    const int qTile  = blockIdx.x * 4 + wave;
    const bool act   = (qTile * 16 < LSEQ);          // wave-uniform
    const int qBase  = qTile * 16;
    const int mrow   = lane & 15;
    const int laneHi = lane >> 4;
    const size_t headOff = (size_t)(n * NUM_HEADS + h) * LSEQ;

    // Per-thread async-copy assignment: one 16B segment of K and V per tile.
    const int cpKey = threadIdx.x >> 2;          // 0..31
    const int cpSeg = (threadIdx.x & 3) * 8;     // f16 offset of the 16B segment
    const unsigned ldsK = (unsigned)(unsigned long long)&sK[cpKey][cpSeg];
    const unsigned ldsV = (unsigned)(unsigned long long)&sV[cpKey][cpSeg];

    // Q fragment (A layout, 16x32) — loaded once
    v16h aq = {};
    if (act) {
        const f16* qrow = qws + (headOff + qBase + mrow) * HEAD_C;
        #pragma unroll
        for (int v = 0; v < 8; ++v) {
            int k = laneHi * 16 + 2 * v;
            aq[2 * v]     = qrow[k];
            aq[2 * v + 1] = qrow[k + 1];
        }
    }

    float runMax[8], runSum[8];
    v8f o0 = {}, o1 = {};
    #pragma unroll
    for (int r = 0; r < 8; ++r) { runMax[r] = -3.0e38f; runSum[r] = 0.0f; }

    const int numKT = (LSEQ + 31) / 32;   // 63
    for (int kt = 0; kt < numKT; ++kt) {
        const int keyBase = kt * 32;

        // --- async stage of K/V tile directly into LDS -------------------
        {
            int gk  = keyBase + cpKey;
            int gkc = gk < LSEQ ? gk : (LSEQ - 1);   // clamp; masked via P==0 below
            const f16* gK = kws + (headOff + gkc) * HEAD_C + cpSeg;
            const f16* gV = vws + (headOff + gkc) * HEAD_C + cpSeg;
            asm volatile("global_load_async_to_lds_b128 %0, %1, off"
                         :: "v"(ldsK), "v"(gK) : "memory");
            asm volatile("global_load_async_to_lds_b128 %0, %1, off"
                         :: "v"(ldsV), "v"(gV) : "memory");
            // prefetch next tile into the cache hierarchy while we compute
            if (gk + 32 < LSEQ) {
                __builtin_prefetch(gK + 32 * HEAD_C, 0, 3);
                __builtin_prefetch(gV + 32 * HEAD_C, 0, 3);
            }
            asm volatile("s_wait_asynccnt 0x0" ::: "memory");
        }
        __syncthreads();

        if (act) {
            // B fragments of K^T: B[d][keycol] = K[key][d]
            v16h bk0, bk1;
            #pragma unroll
            for (int v = 0; v < 8; ++v) {
                int d = laneHi * 16 + 2 * v;
                bk0[2 * v]     = sK[mrow][d];
                bk0[2 * v + 1] = sK[mrow][d + 1];
                bk1[2 * v]     = sK[mrow + 16][d];
                bk1[2 * v + 1] = sK[mrow + 16][d + 1];
            }
            v8f z = {};
            v8f s0 = __builtin_amdgcn_wmma_f32_16x16x32_f16(false, aq, false, bk0,
                                                            (short)0, z, false, false);
            v8f s1 = __builtin_amdgcn_wmma_f32_16x16x32_f16(false, aq, false, bk1,
                                                            (short)0, z, false, false);

            // mask out-of-range key columns (only upper half of the last tile)
            if (keyBase + mrow >= LSEQ) {
                #pragma unroll
                for (int r = 0; r < 8; ++r) s0[r] = -1.0e30f;
            }
            if (keyBase + 16 + mrow >= LSEQ) {
                #pragma unroll
                for (int r = 0; r < 8; ++r) s1[r] = -1.0e30f;
            }

            // per-row max across the 32 keys (rows live per 16-lane half)
            float tmax[8];
            #pragma unroll
            for (int r = 0; r < 8; ++r) tmax[r] = fmaxf(s0[r], s1[r]);
            #pragma unroll
            for (int off = 1; off < 16; off <<= 1) {
                #pragma unroll
                for (int r = 0; r < 8; ++r)
                    tmax[r] = fmaxf(tmax[r], __shfl_xor(tmax[r], off, 32));
            }

            float corr[8];
            #pragma unroll
            for (int r = 0; r < 8; ++r) {
                float nm = fmaxf(runMax[r], tmax[r]);
                corr[r]  = __expf(runMax[r] - nm);
                runMax[r] = nm;
            }

            float p0[8], p1[8], tsum[8];
            #pragma unroll
            for (int r = 0; r < 8; ++r) {
                p0[r] = __expf(s0[r] - runMax[r]);
                p1[r] = __expf(s1[r] - runMax[r]);
                tsum[r] = p0[r] + p1[r];
            }
            #pragma unroll
            for (int off = 1; off < 16; off <<= 1) {
                #pragma unroll
                for (int r = 0; r < 8; ++r)
                    tsum[r] += __shfl_xor(tsum[r], off, 32);
            }
            #pragma unroll
            for (int r = 0; r < 8; ++r) {
                runSum[r] = runSum[r] * corr[r] + tsum[r];
                o0[r] *= corr[r];
                o1[r] *= corr[r];
            }

            // D-layout P -> LDS -> A-layout fragment (same wave, DS in-order)
            #pragma unroll
            for (int r = 0; r < 8; ++r) {
                int m = r + laneHi * 8;
                sP[wave][m][mrow]      = (f16)p0[r];
                sP[wave][m][16 + mrow] = (f16)p1[r];
            }
            v16h ap;
            #pragma unroll
            for (int v = 0; v < 8; ++v) {
                int k = laneHi * 16 + 2 * v;
                ap[2 * v]     = sP[wave][mrow][k];
                ap[2 * v + 1] = sP[wave][mrow][k + 1];
            }

            // B fragments of V: B[keyrow][dcol] = V[key][d]
            v16h bv0, bv1;
            #pragma unroll
            for (int v = 0; v < 8; ++v) {
                int key = laneHi * 16 + 2 * v;
                bv0[2 * v]     = sV[key][mrow];
                bv0[2 * v + 1] = sV[key + 1][mrow];
                bv1[2 * v]     = sV[key][16 + mrow];
                bv1[2 * v + 1] = sV[key + 1][16 + mrow];
            }
            o0 = __builtin_amdgcn_wmma_f32_16x16x32_f16(false, ap, false, bv0,
                                                        (short)0, o0, false, false);
            o1 = __builtin_amdgcn_wmma_f32_16x16x32_f16(false, ap, false, bv1,
                                                        (short)0, o1, false, false);
        }
        __syncthreads();
    }

    if (act) {
        #pragma unroll
        for (int r = 0; r < 8; ++r) {
            float inv = 1.0f / runSum[r];
            int m = r + laneHi * 8;
            int q = qBase + m;
            size_t base = ((size_t)n * LSEQ + q) * OUT_C + h * HEAD_C;
            aws[base + mrow]      = (f16)(o0[r] * inv);
            aws[base + 16 + mrow] = (f16)(o1[r] * inv);
        }
    }
}

// ---------------------------------------------------------------------------
// Kernel 3: out = attn_out @ w_out + b_out  (f32 output); 16x32 per wave.
// ---------------------------------------------------------------------------
__global__ __launch_bounds__(128)
void out_gemm_kernel(const f16* __restrict__ aws, const float* __restrict__ w_out,
                     const float* __restrict__ b_out, float* __restrict__ out)
{
    const int lane    = threadIdx.x & 31;
    const int wave    = threadIdx.x >> 5;
    const int colBase = blockIdx.x * 32;         // 0,32,..,224
    const int rowTile = blockIdx.y * 4 + wave;   // 0..1999
    const int rowBase = rowTile * 16;
    const int mrow    = lane & 15;
    const int laneHi  = lane >> 4;

    v8f acc0 = {}, acc1 = {};
    #pragma unroll
    for (int k0 = 0; k0 < OUT_C; k0 += 32) {
        v16h a, b0, b1;
        const f16* arow = aws + (size_t)(rowBase + mrow) * OUT_C + k0;
        #pragma unroll
        for (int v = 0; v < 8; ++v) {
            int k = laneHi * 16 + 2 * v;
            a[2 * v]     = arow[k];
            a[2 * v + 1] = arow[k + 1];
        }
        #pragma unroll
        for (int v = 0; v < 8; ++v) {
            int k = laneHi * 16 + 2 * v;
            const float* brow0 = w_out + (size_t)(k0 + k)     * OUT_C + colBase + mrow;
            const float* brow1 = w_out + (size_t)(k0 + k + 1) * OUT_C + colBase + mrow;
            b0[2 * v]     = (f16)brow0[0];
            b0[2 * v + 1] = (f16)brow1[0];
            b1[2 * v]     = (f16)brow0[16];
            b1[2 * v + 1] = (f16)brow1[16];
        }
        acc0 = __builtin_amdgcn_wmma_f32_16x16x32_f16(false, a, false, b0,
                                                      (short)0, acc0, false, false);
        acc1 = __builtin_amdgcn_wmma_f32_16x16x32_f16(false, a, false, b1,
                                                      (short)0, acc1, false, false);
    }

    const float bias0 = b_out[colBase + mrow];
    const float bias1 = b_out[colBase + 16 + mrow];
    #pragma unroll
    for (int r = 0; r < 8; ++r) {
        int m = r + laneHi * 8;
        float* orow = out + (size_t)(rowBase + m) * OUT_C + colBase + mrow;
        orow[0]  = acc0[r] + bias0;
        orow[16] = acc1[r] + bias1;
    }
}

// ---------------------------------------------------------------------------
extern "C" void kernel_launch(void* const* d_in, const int* in_sizes, int n_in,
                              void* d_out, int out_size, void* d_ws, size_t ws_size,
                              hipStream_t stream)
{
    const float* x     = (const float*)d_in[0];
    // d_in[1] = cross_x, unused by the reference forward
    const float* w_qkv = (const float*)d_in[2];
    const float* w_out = (const float*)d_in[3];
    const float* b_out = (const float*)d_in[4];

    char* ws = (char*)d_ws;
    const size_t perTensor = (size_t)NBATCH * NUM_HEADS * LSEQ * HEAD_C * sizeof(f16);
    f16* qws = (f16*)(ws);
    f16* kws = (f16*)(ws + perTensor);
    f16* vws = (f16*)(ws + 2 * perTensor);
    f16* aws = (f16*)(ws + 3 * perTensor);   // [N*L, OUT_C] f16

    dim3 blk(128);
    // 32000 rows -> 2000 row tiles -> 500 blocks of 4 waves; 2 col tiles/wave
    qkv_gemm_kernel<<<dim3(QKV_COLS / 32, 500), blk, 0, stream>>>(x, w_qkv, qws, kws, vws);
    // 125 q-tiles per (n,h) -> 32 blocks of 4 waves
    attn_kernel<<<dim3(32, NUM_HEADS, NBATCH), blk, 0, stream>>>(qws, kws, vws, aws);
    out_gemm_kernel<<<dim3(OUT_C / 32, 500), blk, 0, stream>>>(aws, w_out, b_out, (float*)d_out);
}
